// Attention_91345364451520
// MI455X (gfx1250) — 8-bit matrix operand order repaired and checked by reading, compiled, not run on hardware
//
#include <hip/hip_runtime.h>
#include <stdint.h>


typedef __attribute__((ext_vector_type(8))) int v8i;
typedef __attribute__((ext_vector_type(4))) int v4i;
typedef __attribute__((ext_vector_type(4))) unsigned int v4u;

#define NH  12
#define HD  64
#define TOK 4096
#define DIM 768

__device__ __forceinline__ v8i v8i_zero() {
  v8i z = {0, 0, 0, 0, 0, 0, 0, 0};
  return z;
}

__device__ __forceinline__ v8i wmma_iu8(v8i a, v8i b, v8i c) {
  return __builtin_amdgcn_wmma_i32_16x16x64_iu8(true, a, true, b, c, false, false);
}

__device__ __forceinline__ v8i load_a_row64(const int8_t* p, int hi) {
  const int8_t* q = p + (hi ? 8 : 0);
  v8i r;
  r[0] = *(const int*)(q + 0);
  r[1] = *(const int*)(q + 4);
  r[2] = *(const int*)(q + 16);
  r[3] = *(const int*)(q + 20);
  r[4] = *(const int*)(q + 32);
  r[5] = *(const int*)(q + 36);
  r[6] = *(const int*)(q + 48);
  r[7] = *(const int*)(q + 52);
  return r;
}

__device__ __forceinline__ v8i load_b_col64(const int8_t* p, int hi) {
  const int* q = (const int*)(p + hi * 8);
  v8i r;
#pragma unroll
  for (int j = 0; j < 8; ++j) r[j] = q[(j >> 1) * 4 + (j & 1)];
  return r;
}

__device__ __forceinline__ void tdm_copy_1d(uint32_t lds_off, const void* gsrc,
                                            uint32_t nbytes) {
  uint64_t ga = (uint64_t)(uintptr_t)gsrc;
  v4u g0;
  g0[0] = 1u;
  g0[1] = lds_off;
  g0[2] = (uint32_t)ga;
  g0[3] = (uint32_t)((ga >> 32) & 0x01FFFFFFu) | (2u << 30);
  v8i g1;
  g1[0] = 0;
  g1[1] = (int)((nbytes & 0xFFFFu) << 16);
  g1[2] = (int)((nbytes >> 16) & 0xFFFFu) | (1 << 16);
  g1[3] = (int)((nbytes & 0xFFFFu) << 16);
  g1[4] = 0;
  g1[5] = (int)nbytes;
  g1[6] = 0;
  g1[7] = 0;
  v4i z4 = {0, 0, 0, 0};
  v8i z8 = v8i_zero();
  __builtin_amdgcn_tensor_load_to_lds(g0, g1, z4, z4, z8, 0);
}

__global__ void pack_i32_to_i8(const int* __restrict__ src,
                               int8_t* __restrict__ dst, int n) {
  for (int i = blockIdx.x * blockDim.x + threadIdx.x; i < n;
       i += gridDim.x * blockDim.x)
    dst[i] = (int8_t)src[i];
}

__global__ __launch_bounds__(256) void qkv_gemm_kernel(
    const int8_t* __restrict__ x8, const int8_t* __restrict__ w8,
    const int* __restrict__ qkv_b, const float* __restrict__ a_scale_p,
    const float* __restrict__ b_scale_p, int8_t* __restrict__ qb,
    int8_t* __restrict__ kbuf, int8_t* __restrict__ vtb) {
  const int nblk = blockIdx.x, mblk = blockIdx.y;
  const int tid = threadIdx.x, wv = tid >> 5, lane = tid & 31;
  const int l15 = lane & 15, hi = lane >> 4;
  const int mt = wv & 3, nc0 = (wv >> 2) * 64;
  const int row_base = mblk * 64 + mt * 16;
  const int col_base = nblk * 128 + nc0;
  const float a_scale = a_scale_p[0], b_scale = b_scale_p[0];

  v8i acc[4];
#pragma unroll
  for (int t = 0; t < 4; ++t) acc[t] = v8i_zero();

  const int8_t* arow = x8 + (size_t)(row_base + l15) * DIM;

  for (int kb = 0; kb < 12; ++kb) {
    if (kb + 1 < 12) __builtin_prefetch(arow + (kb + 1) * 64, 0, 1);
    v8i a = load_a_row64(arow + kb * 64, hi);
#pragma unroll
    for (int t = 0; t < 4; ++t) {
      const int8_t* bcol =
          w8 + (size_t)(col_base + t * 16 + l15) * DIM + kb * 64;
      v8i b = load_b_col64(bcol, hi);
      acc[t] = wmma_iu8(a, b, acc[t]);
    }
  }

#pragma unroll
  for (int t = 0; t < 4; ++t) {
#pragma unroll
    for (int r = 0; r < 8; ++r) {
      int m = row_base + r + hi * 8;
      int ng = col_base + t * 16 + l15;
      float f = (float)acc[t][r] * a_scale + (float)qkv_b[ng] * b_scale;
      f = fminf(fmaxf(rintf(f), -128.f), 127.f);
      int8_t q8 = (int8_t)(int)f;
      int s = ng / DIM, rem = ng - s * DIM;
      int head = rem >> 6, d = rem & 63;
      if (s == 0)
        qb[((size_t)head * TOK + m) * HD + d] = q8;
      else if (s == 1)
        kbuf[((size_t)head * TOK + m) * HD + d] = q8;
      else
        vtb[((size_t)head * HD + d) * TOK + m] = q8;
    }
  }
}

__global__ __launch_bounds__(128) void rel_gemm_kernel(
    const int8_t* __restrict__ qb, const int8_t* __restrict__ table8,
    const float* __restrict__ scale_p, float* __restrict__ out) {
  const int h = blockIdx.x, head = blockIdx.y;
  const int tid = threadIdx.x, wv = tid >> 5, lane = tid & 31;
  const int l15 = lane & 15, hi = lane >> 4;
  const float scale = scale_p[0];
  const int w0 = wv * 16;
  const int token = h * 64 + w0 + l15;
  v8i a = load_a_row64(qb + ((size_t)head * TOK + token) * HD, hi);
#pragma unroll
  for (int t = 0; t < 4; ++t) {
    int k = t * 16 + l15;
    v8i b = load_b_col64(table8 + (size_t)(h - k + 63) * HD, hi);
    v8i c = wmma_iu8(a, b, v8i_zero());
#pragma unroll
    for (int r = 0; r < 8; ++r) {
      int m = w0 + r + hi * 8;
      out[(((size_t)head * 64 + h) * 64 + m) * 64 + k] = (float)c[r] * scale;
    }
  }
}

__global__ __launch_bounds__(256) void attn_kernel(
    const int8_t* __restrict__ qb, const int8_t* __restrict__ kbuf,
    const int8_t* __restrict__ vtb, const float* __restrict__ relh,
    const float* __restrict__ relw, const float* __restrict__ qk_scale_p,
    const float* __restrict__ pv_scale_p, int8_t* __restrict__ ob) {
  __shared__ int8_t p_lds[16][4096];
  __shared__ float relh_s[16][64];
  __shared__ float relw_s[16][64];
  __shared__ float red_m[8][16];
  __shared__ float red_s[8][16];
  __shared__ int acc_red[8][16 * 64];

  const int tile = blockIdx.x, head = blockIdx.y;
  const int token0 = tile * 16;
  const int hq = token0 >> 6, w0 = token0 & 63;
  const int tid = threadIdx.x, wv = tid >> 5, lane = tid & 31;
  const int l15 = lane & 15, hi = lane >> 4;
  const float qk_scale = qk_scale_p[0], pv_scale = pv_scale_p[0];

  {
    size_t rel_base = (((size_t)head * 64 + hq) * 64 + w0) * 64;
    if (wv == 0) {
      tdm_copy_1d((uint32_t)(uintptr_t)(&relh_s[0][0]), relh + rel_base,
                  16 * 64 * 4);
      tdm_copy_1d((uint32_t)(uintptr_t)(&relw_s[0][0]), relw + rel_base,
                  16 * 64 * 4);
    }
  }

  v8i aq = load_a_row64(qb + ((size_t)head * TOK + token0 + l15) * HD, hi);

  if (wv == 0) __builtin_amdgcn_s_wait_tensorcnt(0);
  __syncthreads();

  const int key_lo = wv * 512;

  float m8[8], s8[8];
#pragma unroll
  for (int r = 0; r < 8; ++r) { m8[r] = -1e30f; s8[r] = 0.f; }

  for (int kb = 0; kb < 32; ++kb) {
    int key0 = key_lo + kb * 16;
    v8i b = load_b_col64(kbuf + ((size_t)head * TOK + key0 + l15) * HD, hi);
    v8i c = wmma_iu8(aq, b, v8i_zero());
    int key = key0 + l15;
    int kh = key >> 6, kw = key & 63;
#pragma unroll
    for (int r = 0; r < 8; ++r) {
      int m = r + hi * 8;
      float f = (float)c[r] * qk_scale + relh_s[m][kh] + relw_s[m][kw];
      float bm = f;
      bm = fmaxf(bm, __shfl_xor(bm, 1, 16));
      bm = fmaxf(bm, __shfl_xor(bm, 2, 16));
      bm = fmaxf(bm, __shfl_xor(bm, 4, 16));
      bm = fmaxf(bm, __shfl_xor(bm, 8, 16));
      float nm = fmaxf(m8[r], bm);
      float e = __expf(f - nm);
      e += __shfl_xor(e, 1, 16);
      e += __shfl_xor(e, 2, 16);
      e += __shfl_xor(e, 4, 16);
      e += __shfl_xor(e, 8, 16);
      s8[r] = s8[r] * __expf(m8[r] - nm) + e;
      m8[r] = nm;
    }
  }
  if (l15 == 0) {
#pragma unroll
    for (int r = 0; r < 8; ++r) {
      red_m[wv][r + hi * 8] = m8[r];
      red_s[wv][r + hi * 8] = s8[r];
    }
  }
  __syncthreads();

  float Mg[8], Sinv[8];
#pragma unroll
  for (int r = 0; r < 8; ++r) {
    int m = r + hi * 8;
    float M = -1e30f;
    for (int w = 0; w < 8; ++w) M = fmaxf(M, red_m[w][m]);
    float S = 0.f;
    for (int w = 0; w < 8; ++w) S += red_s[w][m] * __expf(red_m[w][m] - M);
    Mg[r] = M;
    Sinv[r] = 127.f / S;
  }

  for (int kb = 0; kb < 32; ++kb) {
    int key0 = key_lo + kb * 16;
    v8i b = load_b_col64(kbuf + ((size_t)head * TOK + key0 + l15) * HD, hi);
    v8i c = wmma_iu8(aq, b, v8i_zero());
    int key = key0 + l15;
    int kh = key >> 6, kw = key & 63;
#pragma unroll
    for (int r = 0; r < 8; ++r) {
      int m = r + hi * 8;
      float f = (float)c[r] * qk_scale + relh_s[m][kh] + relw_s[m][kw];
      int pi = (int)rintf(__expf(f - Mg[r]) * Sinv[r]);
      pi = pi < 0 ? 0 : (pi > 127 ? 127 : pi);
      p_lds[m][key] = (int8_t)pi;
    }
  }
  __syncthreads();

  v8i accv[4];
#pragma unroll
  for (int t = 0; t < 4; ++t) accv[t] = v8i_zero();
  for (int kc = 0; kc < 8; ++kc) {
    int key0 = key_lo + kc * 64;
    v8i ap = load_a_row64((const int8_t*)&p_lds[l15][key0], hi);
#pragma unroll
    for (int t = 0; t < 4; ++t) {
      int d = t * 16 + l15;
      v8i bv = load_b_col64(vtb + ((size_t)head * HD + d) * TOK + key0, hi);
      accv[t] = wmma_iu8(ap, bv, accv[t]);
    }
  }
#pragma unroll
  for (int t = 0; t < 4; ++t)
#pragma unroll
    for (int r = 0; r < 8; ++r)
      acc_red[wv][(r + hi * 8) * 64 + t * 16 + l15] = accv[t][r];
  __syncthreads();

  for (int i = tid; i < 16 * 64; i += 256) {
    int m = i >> 6, d = i & 63;
    int sum = 0;
#pragma unroll
    for (int w = 0; w < 8; ++w) sum += acc_red[w][i];
    float f = fminf(fmaxf(rintf((float)sum * pv_scale), -128.f), 127.f);
    ob[(size_t)(token0 + m) * DIM + head * HD + d] = (int8_t)(int)f;
  }
}

__global__ __launch_bounds__(256) void proj_gemm_kernel(
    const int8_t* __restrict__ o8, const int8_t* __restrict__ pw8,
    const float* __restrict__ proj_b, const float* __restrict__ a_scale_p,
    float* __restrict__ out) {
  const int nblk = blockIdx.x, mblk = blockIdx.y;
  const int tid = threadIdx.x, wv = tid >> 5, lane = tid & 31;
  const int l15 = lane & 15, hi = lane >> 4;
  const int mt = wv & 3, nc0 = (wv >> 2) * 64;
  const int row_base = mblk * 64 + mt * 16;
  const int col_base = nblk * 128 + nc0;
  const float a_scale = a_scale_p[0];

  v8i acc[4];
#pragma unroll
  for (int t = 0; t < 4; ++t) acc[t] = v8i_zero();

  const int8_t* arow = o8 + (size_t)(row_base + l15) * DIM;

  for (int kb = 0; kb < 12; ++kb) {
    if (kb + 1 < 12) __builtin_prefetch(arow + (kb + 1) * 64, 0, 1);
    v8i a = load_a_row64(arow + kb * 64, hi);
#pragma unroll
    for (int t = 0; t < 4; ++t) {
      const int8_t* bcol =
          pw8 + (size_t)(col_base + t * 16 + l15) * DIM + kb * 64;
      v8i b = load_b_col64(bcol, hi);
      acc[t] = wmma_iu8(a, b, acc[t]);
    }
  }

#pragma unroll
  for (int t = 0; t < 4; ++t) {
#pragma unroll
    for (int r = 0; r < 8; ++r) {
      int m = row_base + r + hi * 8;
      int ng = col_base + t * 16 + l15;
      out[(size_t)m * DIM + ng] = (float)acc[t][r] * a_scale + proj_b[ng];
    }
  }
}

static inline size_t align256(size_t x) { return (x + 255) & ~(size_t)255; }

extern "C" void kernel_launch(void* const* d_in, const int* in_sizes, int n_in,
                              void* d_out, int out_size, void* d_ws,
                              size_t ws_size, hipStream_t stream) {
  const int* x_i = (const int*)d_in[0];
  const int* qkvw_i = (const int*)d_in[1];
  const int* qkvb_i = (const int*)d_in[2];
  const int* projw_i = (const int*)d_in[3];
  const float* projb = (const float*)d_in[4];
  const int* rph_i = (const int*)d_in[5];
  const int* rpw_i = (const int*)d_in[6];
  const float* qkv_a_scale = (const float*)d_in[7];
  const float* qkv_b_scale = (const float*)d_in[8];
  const float* qk_scale = (const float*)d_in[9];
  const float* rel_h_scale = (const float*)d_in[10];
  const float* rel_w_scale = (const float*)d_in[11];
  const float* pv_scale = (const float*)d_in[12];
  const float* proj_a_scale = (const float*)d_in[13];

  char* ws = (char*)d_ws;
  size_t off = 0;
  auto take = [&](size_t bytes) {
    char* p = ws + off;
    off += align256(bytes);
    return p;
  };
  int8_t* x8 = (int8_t*)take((size_t)TOK * DIM);
  int8_t* w8 = (int8_t*)take((size_t)3 * DIM * DIM);
  int8_t* pw8 = (int8_t*)take((size_t)DIM * DIM);
  int8_t* rh8 = (int8_t*)take((size_t)127 * HD);
  int8_t* rw8 = (int8_t*)take((size_t)127 * HD);
  int8_t* qb = (int8_t*)take((size_t)NH * TOK * HD);
  int8_t* kbuf = (int8_t*)take((size_t)NH * TOK * HD);
  int8_t* vtb = (int8_t*)take((size_t)NH * HD * TOK);
  int8_t* ob = (int8_t*)take((size_t)TOK * DIM);
  float* relh = (float*)take((size_t)NH * 64 * 64 * 64 * 4);
  float* relw = (float*)take((size_t)NH * 64 * 64 * 64 * 4);

  pack_i32_to_i8<<<2048, 256, 0, stream>>>(x_i, x8, TOK * DIM);
  pack_i32_to_i8<<<2048, 256, 0, stream>>>(qkvw_i, w8, 3 * DIM * DIM);
  pack_i32_to_i8<<<1024, 256, 0, stream>>>(projw_i, pw8, DIM * DIM);
  pack_i32_to_i8<<<32, 256, 0, stream>>>(rph_i, rh8, 127 * HD);
  pack_i32_to_i8<<<32, 256, 0, stream>>>(rpw_i, rw8, 127 * HD);

  qkv_gemm_kernel<<<dim3(18, 64), 256, 0, stream>>>(
      x8, w8, qkvb_i, qkv_a_scale, qkv_b_scale, qb, kbuf, vtb);

  rel_gemm_kernel<<<dim3(64, NH), 128, 0, stream>>>(qb, rh8, rel_h_scale,
                                                    relh);
  rel_gemm_kernel<<<dim3(64, NH), 128, 0, stream>>>(qb, rw8, rel_w_scale,
                                                    relw);

  attn_kernel<<<dim3(256, NH), 256, 0, stream>>>(
      qb, kbuf, vtb, relh, relw, qk_scale, pv_scale, ob);

  proj_gemm_kernel<<<dim3(6, 64), 256, 0, stream>>>(ob, pw8, projb,
                                                    proj_a_scale, (float*)d_out);
}
